// SelfAttention_77867757076836
// MI455X (gfx1250) — matrix operand order repaired and checked by reading, not compiled, not run on hardware
//
#include <hip/hip_runtime.h>

// ---------------------------------------------------------------------------
// Flash-attention forward, causal, for MI455X (gfx1250, wave32, WMMA bf16).
// qkv: [b, t, h, 3*d] fp32; key_padding_mask: [b, t] bool; out: [b, t, h, d] fp32
// b=2, t=2048, h=16, d=128.
// ---------------------------------------------------------------------------

typedef __attribute__((ext_vector_type(16))) __bf16 v16bf;
typedef __attribute__((ext_vector_type(8)))  float  v8f;

#define B_    2
#define T_    2048
#define H_    16
#define D_    128
#define NEGV  (-10000.0f)
#define SCALE 0.08838834764831845f   // 1/sqrt(128)

union Frag {
  v16bf  bf;
  __bf16 e[16];
  uint4  q[2];
};

union Pack4 {
  __bf16 b[4];
  uint2  u2;
};

__device__ __forceinline__ v8f zero8() {
  v8f z = {0.f, 0.f, 0.f, 0.f, 0.f, 0.f, 0.f, 0.f};
  return z;
}

__global__ __launch_bounds__(128)
void fa_fwd_causal_kernel(const float* __restrict__ qkv,
                          const unsigned char* __restrict__ kpm,
                          float* __restrict__ out) {
  // Per-wave LDS slice: 4096 bf16 shared K-tile/V^T-tile buffer + 512 bf16 P tile
  __shared__ __align__(16) __bf16 smem[4 * 4608];

  const int lane = threadIdx.x & 31;
  const int wid  = threadIdx.x >> 5;
  const int n16  = lane & 15;   // column index inside a 16-wide tile (N / M select)
  const int hlf  = lane >> 4;   // which half of the wave (selects K-run / M-offset)

  __bf16* sB = smem + wid * 4608; // K tile (32x128 row-major), later V^T (128x32)
  __bf16* sP = sB + 4096;         // P tile, 16 rows x 32 cols, bf16

  const int b  = blockIdx.z;
  const int h  = blockIdx.y;
  const int q0 = blockIdx.x * 64 + wid * 16;   // first query row of this wave's tile

  const size_t tStride = (size_t)H_ * 3 * D_;  // 6144 floats per time step
  const float* Qp = qkv + (size_t)b * T_ * tStride + (size_t)h * 3 * D_;
  const float* Kp = Qp + D_;
  const float* Vp = Qp + 2 * D_;
  const unsigned char* pmRow = kpm + (size_t)b * T_;

  // ---- Preload Q (16 rows x 128 feats) as 4 bf16 A-fragments (K=32 each) ----
  // A layout: lane holds row M=n16; contiguous K-run [hlf*16,+16).
  // 1/sqrt(d) is folded into Q here (S = (Q*scale) K^T), removing per-block muls.
  Frag fq[4];
  #pragma unroll
  for (int kc = 0; kc < 4; ++kc) {
    const float* pa = Qp + (size_t)(q0 + n16) * tStride + kc * 32 + hlf * 16;
    float4 a0 = *(const float4*)(pa);
    float4 a1 = *(const float4*)(pa + 4);
    float4 b0 = *(const float4*)(pa + 8);
    float4 b1 = *(const float4*)(pa + 12);
    fq[kc].e[0]  = (__bf16)(a0.x * SCALE); fq[kc].e[1]  = (__bf16)(a0.y * SCALE);
    fq[kc].e[2]  = (__bf16)(a0.z * SCALE); fq[kc].e[3]  = (__bf16)(a0.w * SCALE);
    fq[kc].e[4]  = (__bf16)(a1.x * SCALE); fq[kc].e[5]  = (__bf16)(a1.y * SCALE);
    fq[kc].e[6]  = (__bf16)(a1.z * SCALE); fq[kc].e[7]  = (__bf16)(a1.w * SCALE);
    fq[kc].e[8]  = (__bf16)(b0.x * SCALE); fq[kc].e[9]  = (__bf16)(b0.y * SCALE);
    fq[kc].e[10] = (__bf16)(b0.z * SCALE); fq[kc].e[11] = (__bf16)(b0.w * SCALE);
    fq[kc].e[12] = (__bf16)(b1.x * SCALE); fq[kc].e[13] = (__bf16)(b1.y * SCALE);
    fq[kc].e[14] = (__bf16)(b1.z * SCALE); fq[kc].e[15] = (__bf16)(b1.w * SCALE);
  }

  // ---- Flash accumulators: O (16x128 f32, 8 C-tiles), row max m, row sum l ----
  v8f acc[8];
  float m_r[8], l_r[8];
  #pragma unroll
  for (int c = 0; c < 8; ++c) acc[c] = zero8();
  #pragma unroll
  for (int r = 0; r < 8; ++r) { m_r[r] = -1e30f; l_r[r] = 0.0f; }

  const int kend = q0 + 16;                    // causal: keys <= last query row
  for (int kb = 0; kb < kend; kb += 32) {
    // ---- Stage K block (32 keys x 128 feats) -> LDS bf16, row-major ----
    // Coalesced: per key row, 32 lanes load 128 contiguous floats.
    #pragma unroll 4
    for (int i = 0; i < 32; ++i) {
      float4 kx = *(const float4*)(Kp + (size_t)(kb + i) * tStride + lane * 4);
      Pack4 pk;
      pk.b[0] = (__bf16)kx.x;
      pk.b[1] = (__bf16)kx.y;
      pk.b[2] = (__bf16)kx.z;
      pk.b[3] = (__bf16)kx.w;
      *(uint2*)(sB + i * 128 + lane * 4) = pk.u2;
    }
    if (kb + 32 < kend) {   // hint the next block into cache
      __builtin_prefetch(Kp + (size_t)(kb + 32) * tStride + lane * 4, 0, 3);
      __builtin_prefetch(Vp + (size_t)(kb + 32) * tStride + lane * 4, 0, 3);
    }
    asm volatile("s_wait_dscnt 0x0" ::: "memory");   // K stores visible to frag loads

    // ---- S = (Q*scale) * K^T : two 16x16 f32 tiles over this 32-key block ----
    // B layout: lane holds column N=n16 (key), contiguous K-run [hlf*16,+16) (feature).
    v8f s0 = zero8(), s1 = zero8();
    #pragma unroll
    for (int kc = 0; kc < 4; ++kc) {
      Frag kb0, kb1;
      const __bf16* r0 = sB + n16 * 128 + kc * 32 + hlf * 16;
      const __bf16* r1 = sB + (16 + n16) * 128 + kc * 32 + hlf * 16;
      kb0.q[0] = *(const uint4*)(r0);
      kb0.q[1] = *(const uint4*)(r0 + 8);
      kb1.q[0] = *(const uint4*)(r1);
      kb1.q[1] = *(const uint4*)(r1 + 8);
      s0 = __builtin_amdgcn_wmma_f32_16x16x32_bf16(false, fq[kc].bf, false, kb0.bf,
                                                   (short)0, s0, false, false);
      s1 = __builtin_amdgcn_wmma_f32_16x16x32_bf16(false, fq[kc].bf, false, kb1.bf,
                                                   (short)0, s1, false, false);
    }
    asm volatile("" ::: "memory");  // keep K-frag DS loads ordered before V overwrite

    // ---- Stage V block transposed into the SAME buffer: sB[f*32 + key] ----
    #pragma unroll 4
    for (int i = 0; i < 32; ++i) {
      float4 vx = *(const float4*)(Vp + (size_t)(kb + i) * tStride + lane * 4);
      sB[(lane * 4 + 0) * 32 + i] = (__bf16)vx.x;
      sB[(lane * 4 + 1) * 32 + i] = (__bf16)vx.y;
      sB[(lane * 4 + 2) * 32 + i] = (__bf16)vx.z;
      sB[(lane * 4 + 3) * 32 + i] = (__bf16)vx.w;
    }

    // ---- Causal/padding mask + online softmax update ----
    // C layout: lane holds N=n16; VGPR r holds row M = r + 8*hlf.
    const bool pad0 = pmRow[kb + n16] != 0;
    const bool pad1 = pmRow[kb + 16 + n16] != 0;
    float p0[8], p1[8];
    #pragma unroll
    for (int r = 0; r < 8; ++r) {
      const int qrow = q0 + r + 8 * hlf;
      float x0 = s0[r];
      float x1 = s1[r];
      if (pad0 || (kb + n16)      > qrow) x0 = NEGV;
      if (pad1 || (kb + 16 + n16) > qrow) x1 = NEGV;
      float mx = fmaxf(x0, x1);
      mx = fmaxf(mx, __shfl_xor(mx, 8));   // reduce across the 16-lane half
      mx = fmaxf(mx, __shfl_xor(mx, 4));
      mx = fmaxf(mx, __shfl_xor(mx, 2));
      mx = fmaxf(mx, __shfl_xor(mx, 1));
      const float mnew = fmaxf(m_r[r], mx);
      const float corr = __expf(m_r[r] - mnew);
      const float e0 = __expf(x0 - mnew);
      const float e1 = __expf(x1 - mnew);
      float rs = e0 + e1;
      rs += __shfl_xor(rs, 8);
      rs += __shfl_xor(rs, 4);
      rs += __shfl_xor(rs, 2);
      rs += __shfl_xor(rs, 1);
      l_r[r] = l_r[r] * corr + rs;
      m_r[r] = mnew;
      p0[r] = e0;
      p1[r] = e1;
      #pragma unroll
      for (int c = 0; c < 8; ++c) acc[c][r] *= corr;
    }

    // ---- P (C layout) -> LDS -> reload as bf16 A-fragment ----
    #pragma unroll
    for (int r = 0; r < 8; ++r) {
      const int m = r + 8 * hlf;
      sP[m * 32 + n16]      = (__bf16)p0[r];
      sP[m * 32 + 16 + n16] = (__bf16)p1[r];
    }
    asm volatile("s_wait_dscnt 0x0" ::: "memory");

    Frag fp;
    fp.q[0] = *(const uint4*)(sP + n16 * 32 + hlf * 16);
    fp.q[1] = *(const uint4*)(sP + n16 * 32 + hlf * 16 + 8);

    // ---- O += P * V : 8 feature chunks of 16 ----
    // B-frag from V^T: lane holds column N = feature (c*16+n16), K-run = keys [hlf*16,+16).
    #pragma unroll
    for (int c = 0; c < 8; ++c) {
      Frag fv;
      const __bf16* vr = sB + (c * 16 + n16) * 32 + hlf * 16;
      fv.q[0] = *(const uint4*)(vr);
      fv.q[1] = *(const uint4*)(vr + 8);
      acc[c] = __builtin_amdgcn_wmma_f32_16x16x32_bf16(false, fp.bf, false, fv.bf,
                                                       (short)0, acc[c], false, false);
    }
  }

  // ---- Epilogue: O / l -> out[b, q, h, f] ----
  #pragma unroll
  for (int r = 0; r < 8; ++r) {
    const float inv = 1.0f / l_r[r];
    const size_t row = (size_t)b * T_ + (size_t)(q0 + r + 8 * hlf);
    float* op = out + (row * H_ + h) * D_ + n16;
    #pragma unroll
    for (int c = 0; c < 8; ++c) op[c * 16] = acc[c][r] * inv;
  }
}

extern "C" void kernel_launch(void* const* d_in, const int* in_sizes, int n_in,
                              void* d_out, int out_size, void* d_ws, size_t ws_size,
                              hipStream_t stream) {
  (void)in_sizes; (void)n_in; (void)out_size; (void)d_ws; (void)ws_size;
  const float*         qkv = (const float*)d_in[0];
  const unsigned char* kpm = (const unsigned char*)d_in[1];
  float*               out = (float*)d_out;

  dim3 grid(T_ / 64, H_, B_);   // x: 4 q-tiles of 16 rows per WG; y: head; z: batch
  dim3 block(128);              // 4 wave32s, each owning one q-tile
  fa_fwd_causal_kernel<<<grid, block, 0, stream>>>(qkv, kpm, out);
}
